// QRAdaptorLayer_53626961657950
// MI455X (gfx1250) — compile-verified, matrix operand order checked by reading, not run on hardware
//
#include <hip/hip_runtime.h>

typedef __attribute__((ext_vector_type(16))) _Float16 v16h;
typedef __attribute__((ext_vector_type(8)))  _Float16 v8h;
typedef __attribute__((ext_vector_type(4)))  _Float16 v4h;
typedef __attribute__((ext_vector_type(8)))  float    v8f;
typedef __attribute__((ext_vector_type(4)))  float    v4f;
typedef __attribute__((ext_vector_type(4)))  int      v4i;

#define K_TOT 4096   // IN_F
#define N_TOT 4096   // OUT_F
#define BM 128
#define BN 128
#define BK 32
#define LDK 40       // padded LDS K-stride in halves (80B/row -> conflict-free)

__device__ __constant__ float kNF4[16] = {
    -1.0f, -0.6961928009986877f, -0.5250730514526367f, -0.39491748809814453f,
    -0.28444138169288635f, -0.18477343022823334f, -0.09105003625154495f, 0.0f,
    0.07958029955625534f, 0.16093020141124725f, 0.24611230194568634f,
    0.33791524171829224f, 0.44070982933044434f, 0.5626170039176941f,
    0.7229568362236023f, 1.0f };

__global__ __launch_bounds__(256)
void qlora_nf4_wmma_kernel(const float* __restrict__ x,
                           const float* __restrict__ w0,
                           const int*   __restrict__ nf4_idx,
                           const float* __restrict__ scale,
                           const float* __restrict__ zp,
                           float* __restrict__ out)
{
    __shared__ _Float16 sX[BM * LDK];
    __shared__ _Float16 sW[BN * LDK];
    __shared__ float    sLut[16];

    const int tid  = threadIdx.x;
    const int lane = tid & 31;
    const int wave = tid >> 5;

    const int m0 = blockIdx.y * BM;
    const int n0 = blockIdx.x * BN;

    // fused affine LUT: level*scale + zero_point (16 floats, once per block)
    if (tid < 16) sLut[tid] = kNF4[tid] * scale[0] + zp[0];

    // wave tile: 64 rows (M) x 32 cols (N); 2x4 wave grid over the 128x128 block
    const int wm   = (wave & 1) * 64;
    const int wn   = (wave >> 1) * 32;
    const int half = lane >> 4;   // K-half select per WMMA A/B layout
    const int hr   = lane & 15;   // row within 16

    v8f acc[4][2];
#pragma unroll
    for (int i = 0; i < 4; ++i)
#pragma unroll
        for (int j = 0; j < 2; ++j)
            acc[i][j] = (v8f){0.f,0.f,0.f,0.f,0.f,0.f,0.f,0.f};

    for (int kb = 0; kb < K_TOT; kb += BK) {
        __syncthreads();  // protect LDS reuse from previous iter (and sLut on iter 0)

        // ---- stage X tile: 128x32 f32 -> f16 in LDS (coalesced float4 loads) ----
#pragma unroll
        for (int i = 0; i < 4; ++i) {
            int s   = i * 256 + tid;        // 0..1023 tile slots of 4 elems
            int row = s >> 3;               // 0..127
            int c4  = (s & 7) << 2;         // 0,4,...,28
            v4f xv = *reinterpret_cast<const v4f*>(
                x + (size_t)(m0 + row) * K_TOT + kb + c4);
            v4h hx = { (_Float16)xv.x, (_Float16)xv.y,
                       (_Float16)xv.z, (_Float16)xv.w };
            *reinterpret_cast<v4h*>(&sX[row * LDK + c4]) = hx;
        }

        // ---- stage W tile: fused NF4 dequant  w0 + lut[idx]  -> f16 in LDS ----
#pragma unroll
        for (int i = 0; i < 4; ++i) {
            int s   = i * 256 + tid;
            int row = s >> 3;
            int c4  = (s & 7) << 2;
            size_t g = (size_t)(n0 + row) * K_TOT + kb + c4;
            v4f wv = *reinterpret_cast<const v4f*>(w0 + g);
            v4i iv = *reinterpret_cast<const v4i*>(nf4_idx + g);
            v4h hw = { (_Float16)(wv.x + sLut[iv.x & 15]),
                       (_Float16)(wv.y + sLut[iv.y & 15]),
                       (_Float16)(wv.z + sLut[iv.z & 15]),
                       (_Float16)(wv.w + sLut[iv.w & 15]) };
            *reinterpret_cast<v4h*>(&sW[row * LDK + c4]) = hw;
        }
        __syncthreads();

        // ---- warm L2 for next K-block (global_prefetch_b8) ----
        if (kb + BK < K_TOT) {
            int row = tid & 127;
            if (tid < 128) {
                __builtin_prefetch(x + (size_t)(m0 + row) * K_TOT + kb + BK, 0, 1);
            } else {
                __builtin_prefetch(w0 + (size_t)(n0 + row) * K_TOT + kb + BK, 0, 1);
                __builtin_prefetch(nf4_idx + (size_t)(n0 + row) * K_TOT + kb + BK, 0, 1);
            }
        }
        // control flow reconverges here -> EXEC all-1s for WMMA

        // ---- load fragments per ISA 16-bit A/B layout: two 16B runs per lane ----
        v16h a[4], b[2];
#pragma unroll
        for (int i = 0; i < 4; ++i) {
            const _Float16* p = &sX[(wm + i * 16 + hr) * LDK + half * 8];
            v8h lo = *reinterpret_cast<const v8h*>(p);        // K = half*8 .. +7
            v8h hi = *reinterpret_cast<const v8h*>(p + 16);   // K = 16+half*8 .. +7
            a[i] = __builtin_shufflevector(lo, hi,
                     0,1,2,3,4,5,6,7,8,9,10,11,12,13,14,15);
        }
#pragma unroll
        for (int j = 0; j < 2; ++j) {
            const _Float16* p = &sW[(wn + j * 16 + hr) * LDK + half * 8];
            v8h lo = *reinterpret_cast<const v8h*>(p);
            v8h hi = *reinterpret_cast<const v8h*>(p + 16);
            b[j] = __builtin_shufflevector(lo, hi,
                     0,1,2,3,4,5,6,7,8,9,10,11,12,13,14,15);
        }

        // ---- 8x v_wmma_f32_16x16x32_f16, f32 accumulation ----
#pragma unroll
        for (int i = 0; i < 4; ++i)
#pragma unroll
            for (int j = 0; j < 2; ++j)
                acc[i][j] = __builtin_amdgcn_wmma_f32_16x16x32_f16(
                    false, a[i], false, b[j],
                    (short)0, acc[i][j], false, false);
    }

    // ---- epilogue: C/D layout -> lane holds N=lane&15, VGPR v -> M = 8*(lane>>4)+v
#pragma unroll
    for (int i = 0; i < 4; ++i) {
        int gmBase = m0 + wm + i * 16 + half * 8;
#pragma unroll
        for (int j = 0; j < 2; ++j) {
            int gn = n0 + wn + j * 16 + hr;
#pragma unroll
            for (int v = 0; v < 8; ++v)
                out[(size_t)(gmBase + v) * N_TOT + gn] = acc[i][j][v];
        }
    }
}

extern "C" void kernel_launch(void* const* d_in, const int* in_sizes, int n_in,
                              void* d_out, int out_size, void* d_ws, size_t ws_size,
                              hipStream_t stream) {
    const float* x     = (const float*)d_in[0];   // [B,S,IN] f32
    const float* w0    = (const float*)d_in[1];   // [OUT,IN] f32
    const int*   idx   = (const int*)  d_in[2];   // [OUT,IN] i32
    const float* scale = (const float*)d_in[3];   // [1] f32
    const float* zp    = (const float*)d_in[4];   // [1] f32
    float*       out   = (float*)d_out;           // [B,S,OUT] f32

    int M = in_sizes[0] / K_TOT;   // B*S = 1024
    int N = in_sizes[1] / K_TOT;   // OUT = 4096
    dim3 grid(N / BN, M / BM);     // 32 x 8 = 256 workgroups
    qlora_nf4_wmma_kernel<<<grid, dim3(256), 0, stream>>>(x, w0, idx, scale, zp, out);
}
